// RNN_31293131718778
// MI455X (gfx1250) — matrix operand order repaired and checked by reading, not compiled, not run on hardware
//
#include <hip/hip_runtime.h>

// ---------------------------------------------------------------------------
// RNN: h_t^l = tanh(inp @ W_ih^T + b_ih + h_{t-1}^l @ W_hh^T), 4 layers,
// B=256, T=256, I=64, H=2048.  bf16 WMMA (v_wmma_f32_16x16x32_bf16) GEMMs,
// f32 accumulation, weights transposed+converted to bf16 once (56 MB, stays
// resident in the 192 MB L2 across the 256 sequential timesteps).
//
// GEMM kernel: 128 threads (4 waves, wave32), block tile 64(M) x 128(N),
// wave tile 32x64 (8 WMMAs per K-step per wave), K-step 32, double-buffered
// LDS with register-staged global prefetch (one barrier per K-step), step
// loop unrolled x2.  amdgpu_waves_per_eu(1) gives the allocator a full
// register budget so the ~155 live VGPRs (acc+frags+staging) never spill;
// staging uses named scalar uint4s (no memory-backed arrays); segment base
// pointers are hoisted so the inner loop has no 64-bit multiplies.
// ---------------------------------------------------------------------------

typedef __attribute__((ext_vector_type(16))) __bf16 v16bf;
typedef __attribute__((ext_vector_type(8)))  float  v8f;

static constexpr int Bsz = 256;   // batch
static constexpr int Tsz = 256;   // time steps
static constexpr int Isz = 64;    // input size
static constexpr int Hsz = 2048;  // hidden
static constexpr int Lsz = 4;     // layers

static constexpr int MT = 64;     // block tile M
static constexpr int NT = 128;    // block tile N
static constexpr int KT = 32;     // K step
static constexpr int LDA = 40;    // padded bf16 row stride of A tile (16B aligned)
static constexpr int LDB = 136;   // padded bf16 row stride of B tile (16B aligned)

union FragU { v16bf v; uint4 q[2]; };

// --- pipeline stage macros (named scalars only, nothing memory-backed) -----
#define STAGE(stepv)                                                          \
    {                                                                         \
        const int  _s   = (stepv);                                            \
        const bool _sg0 = _s < steps0;                                        \
        const int  _kk  = _sg0 ? _s * KT : (_s - steps0) * KT;                \
        const __bf16* _pa = (_sg0 ? rowA0 : rowA1) + _kk;                     \
        const __bf16* _pb = (_sg0 ? colB0 : colB1) + ((size_t)_kk << 11);     \
        sa0 = ((const uint4*)_pa)[0];                                         \
        sa1 = ((const uint4*)_pa)[1];                                         \
        sb0 = ((const uint4*)_pb)[0];                                         \
        sb1 = ((const uint4*)_pb)[1];                                         \
        sb2 = ((const uint4*)_pb)[2];                                         \
        sb3 = ((const uint4*)_pb)[3];                                         \
    }

#define COMMIT(buf)                                                           \
    {                                                                         \
        uint4* _da = (uint4*)(shA[buf] + ra * LDA + ca);                      \
        _da[0] = sa0; _da[1] = sa1;                                           \
        uint4* _db = (uint4*)(shB[buf] + rb * LDB + cb);                      \
        _db[0] = sb0; _db[1] = sb1; _db[2] = sb2; _db[3] = sb3;               \
    }

#define COMPUTE(buf)                                                          \
    {                                                                         \
        FragU af0, af1, bf0, bf1, bf2, bf3;                                   \
        af0.q[0] = *(const uint4*)(shA[buf] + rowf0 * LDA + g);               \
        af0.q[1] = *(const uint4*)(shA[buf] + rowf0 * LDA + 16 + g);          \
        af1.q[0] = *(const uint4*)(shA[buf] + rowf1 * LDA + g);               \
        af1.q[1] = *(const uint4*)(shA[buf] + rowf1 * LDA + 16 + g);          \
        const __bf16* _pb0 = shB[buf] + colf + (lane & 15);                   \
        _Pragma("unroll")                                                     \
        for (int _e = 0; _e < 16; ++_e) {                                     \
            const int _k = (_e < 8) ? (g + _e) : (16 + g + (_e - 8));         \
            const __bf16* _pk = _pb0 + _k * LDB;                              \
            bf0.v[_e] = _pk[0];  bf1.v[_e] = _pk[16];                         \
            bf2.v[_e] = _pk[32]; bf3.v[_e] = _pk[48];                         \
        }                                                                     \
        acc00 = __builtin_amdgcn_wmma_f32_16x16x32_bf16(false, af0.v, false, bf0.v, (short)0, acc00, false, false); \
        acc01 = __builtin_amdgcn_wmma_f32_16x16x32_bf16(false, af0.v, false, bf1.v, (short)0, acc01, false, false); \
        acc02 = __builtin_amdgcn_wmma_f32_16x16x32_bf16(false, af0.v, false, bf2.v, (short)0, acc02, false, false); \
        acc03 = __builtin_amdgcn_wmma_f32_16x16x32_bf16(false, af0.v, false, bf3.v, (short)0, acc03, false, false); \
        acc10 = __builtin_amdgcn_wmma_f32_16x16x32_bf16(false, af1.v, false, bf0.v, (short)0, acc10, false, false); \
        acc11 = __builtin_amdgcn_wmma_f32_16x16x32_bf16(false, af1.v, false, bf1.v, (short)0, acc11, false, false); \
        acc12 = __builtin_amdgcn_wmma_f32_16x16x32_bf16(false, af1.v, false, bf2.v, (short)0, acc12, false, false); \
        acc13 = __builtin_amdgcn_wmma_f32_16x16x32_bf16(false, af1.v, false, bf3.v, (short)0, acc13, false, false); \
    }

// D = tanh(A0[MxK0]*W0[K0xN] + A1[MxK1]*W1[K1xN] + bias[N]) -> bf16 out [MxN]
__global__ __launch_bounds__(128)
__attribute__((amdgpu_waves_per_eu(1)))
void rnn_cell_gemm(const __bf16* __restrict__ A0, const __bf16* __restrict__ W0,
                   int K0, int sA0,
                   const __bf16* __restrict__ A1, const __bf16* __restrict__ W1,
                   int K1, int sA1,
                   const float* __restrict__ bias, __bf16* __restrict__ out)
{
    __shared__ __attribute__((aligned(16))) __bf16 shA[2][MT * LDA];
    __shared__ __attribute__((aligned(16))) __bf16 shB[2][KT * LDB];

    const int tid  = threadIdx.x;
    const int lane = tid & 31;
    const int wave = tid >> 5;     // 4 waves
    const int wm   = wave & 1;     // 2 waves along M (32 rows each)
    const int wn   = wave >> 1;    // 2 waves along N (64 cols each)
    const int m0   = blockIdx.y * MT;
    const int n0   = blockIdx.x * NT;

    v8f acc00 = v8f{}, acc01 = v8f{}, acc02 = v8f{}, acc03 = v8f{};
    v8f acc10 = v8f{}, acc11 = v8f{}, acc12 = v8f{}, acc13 = v8f{};

    const int steps0 = K0 / KT;
    const int total  = steps0 + K1 / KT;   // always even for this problem

    const int am = lane & 15;         // A row within 16 (ISA 16-bit A layout)
    const int g  = (lane >> 4) * 8;   // A K-group select

    // per-thread staging coordinates
    const int ra = tid >> 1, ca = (tid & 1) * 16;   // A tile: half row (32B)
    const int rb = tid >> 2, cb = (tid & 3) * 32;   // B tile: quarter row (64B)

    // fragment coordinates
    const int rowf0 = wm * 32 + am;        // mf = 0
    const int rowf1 = wm * 32 + 16 + am;   // mf = 1
    const int colf  = wn * 64;             // 4 consecutive 16-col frags

    // hoisted segment base pointers (64-bit multiplies out of the loop)
    const __bf16* rowA0 = A0 + (size_t)(m0 + ra) * sA0 + ca;
    const __bf16* rowA1 = A1 + (size_t)(m0 + ra) * sA1 + ca;
    const __bf16* colB0 = W0 + (size_t)rb * Hsz + n0 + cb;
    const __bf16* colB1 = W1 + (size_t)rb * Hsz + n0 + cb;

    uint4 sa0, sa1, sb0, sb1, sb2, sb3;    // staged K-tile (registers)

    STAGE(0);
    for (int step = 0; step < total; step += 2) {
        const int s1 = (step + 1 < total) ? step + 1 : total - 1;  // clamped
        const int s2 = (step + 2 < total) ? step + 2 : total - 1;

        COMMIT(0);
        __syncthreads();
        STAGE(s1);        // global prefetch overlaps buf0 compute
        COMPUTE(0);

        COMMIT(1);
        __syncthreads();
        STAGE(s2);        // global prefetch overlaps buf1 compute
        COMPUTE(1);
    }

    // epilogue: bias + tanh, store bf16 (C/D layout: VGPR r -> M, lane&15 -> N)
    const int mb0 = m0 + wm * 32 + (lane >> 4) * 8;
    const int nb  = n0 + wn * 64 + (lane & 15);
    v8f accs[2][4] = {{acc00, acc01, acc02, acc03}, {acc10, acc11, acc12, acc13}};
    #pragma unroll
    for (int nf = 0; nf < 4; ++nf) {
        const float bn = bias[nb + nf * 16];
        #pragma unroll
        for (int mf = 0; mf < 2; ++mf) {
            #pragma unroll
            for (int r = 0; r < 8; ++r) {
                float v = tanhf(accs[mf][nf][r] + bn);
                out[(size_t)(mb0 + mf * 16 + r) * Hsz + nb + nf * 16] = (__bf16)v;
            }
        }
    }
}

// --- helpers ---------------------------------------------------------------

__global__ void cvt_f32_bf16(__bf16* __restrict__ dst, const float* __restrict__ src, long n)
{
    long i = (long)blockIdx.x * blockDim.x + threadIdx.x;
    if (i < n) dst[i] = (__bf16)src[i];
}

// dst[k*N + n] = src[n*K + k]  (store weight transposed: [K x N] bf16)
__global__ void transpose_cvt(__bf16* __restrict__ dst, const float* __restrict__ src,
                              int K, int N)
{
    long i = (long)blockIdx.x * blockDim.x + threadIdx.x;
    long total = (long)K * N;
    if (i < total) {
        int k = (int)(i / N);
        int n = (int)(i % N);
        dst[i] = (__bf16)src[(long)n * K + k];
    }
}

__global__ void zero_bf16(__bf16* __restrict__ dst, long n)
{
    long i = (long)blockIdx.x * blockDim.x + threadIdx.x;
    if (i < n) dst[i] = (__bf16)0.0f;
}

// out[b] = sum_h hfin[b][h] * Wfc[h] + bfc[0]  (O == 1)
__global__ __launch_bounds__(256)
void fc_kernel(const __bf16* __restrict__ hfin, const float* __restrict__ Wfc,
               const float* __restrict__ bfc, float* __restrict__ out)
{
    __shared__ float red[256];
    const int b = blockIdx.x;
    float s = 0.0f;
    for (int i = threadIdx.x; i < Hsz; i += 256)
        s += (float)hfin[(size_t)b * Hsz + i] * Wfc[i];
    red[threadIdx.x] = s;
    __syncthreads();
    for (int off = 128; off > 0; off >>= 1) {
        if (threadIdx.x < off) red[threadIdx.x] += red[threadIdx.x + off];
        __syncthreads();
    }
    if (threadIdx.x == 0) out[b] = red[0] + bfc[0];
}

// ---------------------------------------------------------------------------

extern "C" void kernel_launch(void* const* d_in, const int* in_sizes, int n_in,
                              void* d_out, int out_size, void* d_ws, size_t ws_size,
                              hipStream_t stream)
{
    const float* x         = (const float*)d_in[0];  // [B,T,I]
    const float* W_ih0     = (const float*)d_in[1];  // [H,I]
    const float* b_ih0     = (const float*)d_in[2];  // [H]
    const float* W_ih_rest = (const float*)d_in[3];  // [L-1,H,H]
    const float* b_ih_rest = (const float*)d_in[4];  // [L-1,H]
    const float* W_hh      = (const float*)d_in[5];  // [L,H,H]
    const float* W_fc      = (const float*)d_in[6];  // [O,H]
    const float* b_fc      = (const float*)d_in[7];  // [O]
    float* outp = (float*)d_out;

    // workspace layout (bf16 elements)
    __bf16* ws = (__bf16*)d_ws;
    size_t off = 0;
    __bf16* xbf = ws + off; off += (size_t)Bsz * Tsz * Isz;       // x as bf16
    __bf16* W0t = ws + off; off += (size_t)Isz * Hsz;             // W_ih0^T   [I x H]
    __bf16* Wrt = ws + off; off += (size_t)(Lsz - 1) * Hsz * Hsz; // W_ih_rest^T [H x H] x3
    __bf16* Wht = ws + off; off += (size_t)Lsz * Hsz * Hsz;       // W_hh^T    [H x H] x4
    __bf16* hA  = ws + off; off += (size_t)Lsz * Bsz * Hsz;       // h ping
    __bf16* hB  = ws + off; off += (size_t)Lsz * Bsz * Hsz;       // h pong

    const int THR = 256;
    {
        long n = (long)Bsz * Tsz * Isz;
        cvt_f32_bf16<<<(unsigned)((n + THR - 1) / THR), THR, 0, stream>>>(xbf, x, n);
    }
    {
        long n = (long)Isz * Hsz;
        transpose_cvt<<<(unsigned)((n + THR - 1) / THR), THR, 0, stream>>>(W0t, W_ih0, Isz, Hsz);
    }
    for (int l = 0; l < Lsz - 1; ++l) {
        long n = (long)Hsz * Hsz;
        transpose_cvt<<<(unsigned)((n + THR - 1) / THR), THR, 0, stream>>>(
            Wrt + (size_t)l * Hsz * Hsz, W_ih_rest + (size_t)l * Hsz * Hsz, Hsz, Hsz);
    }
    for (int l = 0; l < Lsz; ++l) {
        long n = (long)Hsz * Hsz;
        transpose_cvt<<<(unsigned)((n + THR - 1) / THR), THR, 0, stream>>>(
            Wht + (size_t)l * Hsz * Hsz, W_hh + (size_t)l * Hsz * Hsz, Hsz, Hsz);
    }
    {
        long n = (long)Lsz * Bsz * Hsz;
        zero_bf16<<<(unsigned)((n + THR - 1) / THR), THR, 0, stream>>>(hA, n);
    }

    dim3 grid(Hsz / NT, Bsz / MT);   // 16 x 4 = 64 workgroups, 128 threads each
    __bf16* cur = hA;
    __bf16* nxt = hB;
    const size_t HS = (size_t)Bsz * Hsz;

    for (int t = 0; t < Tsz; ++t) {
        // layer 0: x_t @ W_ih0^T + h0 @ W_hh0^T
        rnn_cell_gemm<<<grid, 128, 0, stream>>>(
            xbf + (size_t)t * Isz, W0t, Isz, Tsz * Isz,
            cur,                  Wht, Hsz, Hsz,
            b_ih0, nxt);
        // layers 1..3
        for (int l = 1; l < Lsz; ++l) {
            rnn_cell_gemm<<<grid, 128, 0, stream>>>(
                nxt + (size_t)(l - 1) * HS, Wrt + (size_t)(l - 1) * Hsz * Hsz, Hsz, Hsz,
                cur + (size_t)l * HS,       Wht + (size_t)l * Hsz * Hsz,       Hsz, Hsz,
                b_ih_rest + (size_t)(l - 1) * Hsz, nxt + (size_t)l * HS);
        }
        __bf16* tmp = cur; cur = nxt; nxt = tmp;
    }

    // final FC: out[b] = h_final[L-1] @ W_fc^T + b_fc
    fc_kernel<<<Bsz, THR, 0, stream>>>(cur + (size_t)(Lsz - 1) * HS, W_fc, b_fc, outp);
}
